// SparseMoELayer_40355512714058
// MI455X (gfx1250) — matrix operand order repaired and checked by reading, not compiled, not run on hardware
//
#include <hip/hip_runtime.h>
#include <hip/hip_bf16.h>

// ---------------- problem constants ----------------
#define NTOK 8192      // B*T
#define DDIM 1024
#define NEXP 8
#define FFD  4096
#define KSEL 2
#define ROWS_TOTAL (NTOK * KSEL)          // 16384 exact assignments
#define TILE_M     32                     // row-tile height (per-expert pad unit)
#define ROWS_PAD   16640                  // 16384 + 8*31 rounded to 32
#define ROW_TILES  (ROWS_PAD / TILE_M)    // 520 worst-case row tiles

typedef __attribute__((ext_vector_type(16))) __bf16 v16bf;
typedef __attribute__((ext_vector_type(8)))  float  v8f;

union AFrag { v16bf v; uint4 q[2]; };

__device__ __forceinline__ unsigned short f2bf(float f) {
  unsigned int u = __float_as_uint(f);
  unsigned int r = u + 0x7FFFu + ((u >> 16) & 1u);   // round-to-nearest-even
  return (unsigned short)(r >> 16);
}

// A-matrix 16x32 bf16 fragment (ISA 7.12.2 lane layout): lane = m + 16h
// elements 0..7  <- K = k0 + 8h .. +7 ; elements 8..15 <- K = k0 + 16 + 8h .. +7
__device__ __forceinline__ void loadA(AFrag& f, const unsigned short* row, int k0, int h) {
  const uint4* p = (const uint4*)(row + k0 + 8 * h);
  f.q[0] = p[0];
  f.q[1] = p[2];
}
// B-matrix 32x16 bf16 fragment: lane = n + 16h, elements 0..7 <- K = k0 + 8h .. +7 ; elements 8..15 <- K = k0 + 16 + 8h .. +7
__device__ __forceinline__ void loadB(AFrag& f, const unsigned short* row, int k0, int h) {
  const uint4* p = (const uint4*)(row + k0 + 8 * h);
  f.q[0] = p[0];
  f.q[1] = p[2];
}
__device__ __forceinline__ v8f wmma_bf16(const AFrag& a, const AFrag& b, v8f c) {
  return __builtin_amdgcn_wmma_f32_16x16x32_bf16(false, a.v, false, b.v,
                                                 (short)0, c, false, false);
}

// ---------------- init: counters + padded row arrays ----------------
__global__ void init_kernel(int* counts, float* sumP, float* sumZ,
                            int* rowToken, float* rowWeight, int* destIdx) {
  int i = blockIdx.x * blockDim.x + threadIdx.x;
  if (i < ROWS_PAD) {
    rowToken[i]  = 0;          // pad rows read token 0 (valid memory)
    rowWeight[i] = 0.0f;       // ...but contribute nothing
    destIdx[i]   = 2 * NTOK;   // dummy output row
  }
  if (i < NEXP) { counts[i] = 0; sumP[i] = 0.0f; }
  if (i == 0)   { sumZ[0] = 0.0f; }
}

// ---------------- fp32 -> bf16 weight conversion ----------------
__global__ void wcvt_kernel(const float* __restrict__ src,
                            unsigned short* __restrict__ dst, int n4) {
  int i = blockIdx.x * blockDim.x + threadIdx.x;
  if (i >= n4) return;
  float4 f = ((const float4*)src)[i];
  ushort4 o;
  o.x = f2bf(f.x); o.y = f2bf(f.y); o.z = f2bf(f.z); o.w = f2bf(f.w);
  ((ushort4*)dst)[i] = o;
}

// ---------------- router: logits, softmax, top-2, aux stats, x->bf16 ----------------
__global__ void __launch_bounds__(256)
router_kernel(const float* __restrict__ x, const float* __restrict__ gateW,
              unsigned short* __restrict__ xb, int* __restrict__ topkIdx,
              float* __restrict__ topkW, int* counts, float* sumP, float* sumZ) {
  __shared__ float red[256 * NEXP];
  int tok = blockIdx.x;
  int t   = threadIdx.x;
  const float* xr = x + (size_t)tok * DDIM;

  float p[NEXP];
#pragma unroll
  for (int e = 0; e < NEXP; ++e) p[e] = 0.0f;

  for (int d = t; d < DDIM; d += 256) {
    float xv = xr[d];
    xb[(size_t)tok * DDIM + d] = f2bf(xv);
#pragma unroll
    for (int e = 0; e < NEXP; ++e) p[e] += xv * gateW[e * DDIM + d];
  }
#pragma unroll
  for (int e = 0; e < NEXP; ++e) red[t * NEXP + e] = p[e];
  __syncthreads();
  for (int s = 128; s > 0; s >>= 1) {
    if (t < s) {
#pragma unroll
      for (int e = 0; e < NEXP; ++e) red[t * NEXP + e] += red[(t + s) * NEXP + e];
    }
    __syncthreads();
  }
  if (t == 0) {
    float l[NEXP], pr[NEXP];
#pragma unroll
    for (int e = 0; e < NEXP; ++e) l[e] = red[e];
    float mx = l[0];
#pragma unroll
    for (int e = 1; e < NEXP; ++e) mx = fmaxf(mx, l[e]);
    float se = 0.0f;
#pragma unroll
    for (int e = 0; e < NEXP; ++e) { pr[e] = __expf(l[e] - mx); se += pr[e]; }
    float inv = 1.0f / se;
#pragma unroll
    for (int e = 0; e < NEXP; ++e) pr[e] *= inv;
    float lse = __logf(se) + mx;

    int i1 = 0;
#pragma unroll
    for (int e = 1; e < NEXP; ++e) if (pr[e] > pr[i1]) i1 = e;
    int i2 = (i1 == 0) ? 1 : 0;
#pragma unroll
    for (int e = 0; e < NEXP; ++e) if (e != i1 && pr[e] > pr[i2]) i2 = e;

    float v1 = pr[i1], v2 = pr[i2];
    float s2 = v1 + v2 + 1e-6f;
    topkIdx[tok * 2 + 0] = i1; topkIdx[tok * 2 + 1] = i2;
    topkW[tok * 2 + 0] = v1 / s2; topkW[tok * 2 + 1] = v2 / s2;

    atomicAdd(&counts[i1], 1);
    atomicAdd(&counts[i2], 1);
#pragma unroll
    for (int e = 0; e < NEXP; ++e) atomicAdd(&sumP[e], pr[e]);
    atomicAdd(sumZ, lse * lse);
  }
}

// ---------------- exclusive padded scan of expert counts ----------------
__global__ void scan_kernel(const int* counts, int* padOff, int* cursor) {
  if (threadIdx.x == 0 && blockIdx.x == 0) {
    int off = 0;
    for (int e = 0; e < NEXP; ++e) {
      padOff[e] = off;
      cursor[e] = off;
      off += ((counts[e] + TILE_M - 1) / TILE_M) * TILE_M;
    }
    padOff[NEXP] = off;
  }
}

// ---------------- scatter tokens into per-expert row segments ----------------
__global__ void scatter_kernel(const int* __restrict__ topkIdx,
                               const float* __restrict__ topkW, int* cursor,
                               int* rowToken, float* rowWeight, int* destIdx) {
  int tok = blockIdx.x * blockDim.x + threadIdx.x;
  if (tok >= NTOK) return;
#pragma unroll
  for (int k = 0; k < KSEL; ++k) {
    int e   = topkIdx[tok * 2 + k];
    int pos = atomicAdd(&cursor[e], 1);
    rowToken[pos]  = tok;
    rowWeight[pos] = topkW[tok * 2 + k];
    destIdx[pos]   = tok * 2 + k;
  }
}

// ---------------- aux loss scalar ----------------
__global__ void aux_kernel(const int* counts, const float* sumP,
                           const float* sumZ, float* out_aux) {
  if (threadIdx.x == 0 && blockIdx.x == 0) {
    float lb = 0.0f;
    for (int e = 0; e < NEXP; ++e)
      lb += ((float)counts[e] / (float)(NTOK * KSEL)) * (sumP[e] / (float)NTOK);
    lb *= (float)NEXP;
    float z = (sumZ[0] / (float)NTOK) * 1e-3f;
    out_aux[0] = 0.01f * (lb + z);
  }
}

// ---------------- FFN stage 1: H = silu(X Wg^T) * (X Wu^T) ----------------
// block = 128 threads (4 wave32). Wave tile: 32 rows x 32 FF for gate AND up.
// Per k-iter: 4 A loads + 8 B loads feed 8 WMMAs (1.5 b128 loads / WMMA).
__global__ void __launch_bounds__(128)
ffn1_kernel(const unsigned short* __restrict__ xb,
            const unsigned short* __restrict__ Wgb,
            const unsigned short* __restrict__ Wub,
            const int* __restrict__ rowToken, const int* __restrict__ padOff,
            unsigned short* __restrict__ H) {
  int rowBase = blockIdx.x * TILE_M;
  if (rowBase >= padOff[NEXP]) return;              // uniform exit: EXEC stays full
  int e = 0;
#pragma unroll
  for (int i = 1; i < NEXP; ++i) if (rowBase >= padOff[i]) e = i;

  int lane = threadIdx.x & 31;
  int wave = threadIdx.x >> 5;
  int m = lane & 15;                                // A row / B,C col index
  int h = lane >> 4;                                // K half select
  int fbase = blockIdx.y * 128 + wave * 32;         // this wave's 32 FF cols

  int tok0 = rowToken[rowBase + m];
  int tok1 = rowToken[rowBase + 16 + m];
  const unsigned short* a0 = xb + (size_t)tok0 * DDIM;
  const unsigned short* a1 = xb + (size_t)tok1 * DDIM;
  const unsigned short* g0 = Wgb + ((size_t)e * FFD + fbase + m) * DDIM;
  const unsigned short* g1 = Wgb + ((size_t)e * FFD + fbase + 16 + m) * DDIM;
  const unsigned short* u0 = Wub + ((size_t)e * FFD + fbase + m) * DDIM;
  const unsigned short* u1 = Wub + ((size_t)e * FFD + fbase + 16 + m) * DDIM;

  v8f ag[2][2], au[2][2];
#pragma unroll
  for (int rr = 0; rr < 2; ++rr)
#pragma unroll
    for (int cc = 0; cc < 2; ++cc) {
      ag[rr][cc] = {0.f, 0.f, 0.f, 0.f, 0.f, 0.f, 0.f, 0.f};
      au[rr][cc] = {0.f, 0.f, 0.f, 0.f, 0.f, 0.f, 0.f, 0.f};
    }

  for (int k0 = 0; k0 < DDIM; k0 += 32) {
    AFrag A0, A1, Bg0, Bg1, Bu0, Bu1;
    loadA(A0, a0, k0, h);
    loadA(A1, a1, k0, h);
    loadB(Bg0, g0, k0, h);
    loadB(Bg1, g1, k0, h);
    loadB(Bu0, u0, k0, h);
    loadB(Bu1, u1, k0, h);
    ag[0][0] = wmma_bf16(A0, Bg0, ag[0][0]);
    ag[0][1] = wmma_bf16(A0, Bg1, ag[0][1]);
    ag[1][0] = wmma_bf16(A1, Bg0, ag[1][0]);
    ag[1][1] = wmma_bf16(A1, Bg1, ag[1][1]);
    au[0][0] = wmma_bf16(A0, Bu0, au[0][0]);
    au[0][1] = wmma_bf16(A0, Bu1, au[0][1]);
    au[1][0] = wmma_bf16(A1, Bu0, au[1][0]);
    au[1][1] = wmma_bf16(A1, Bu1, au[1][1]);
  }
#pragma unroll
  for (int rr = 0; rr < 2; ++rr)
#pragma unroll
    for (int cc = 0; cc < 2; ++cc)
#pragma unroll
      for (int r = 0; r < 8; ++r) {
        int mr = rowBase + rr * 16 + r + 8 * h;     // C/D: VGPR r -> row r + 8*(lane/16)
        int fc = fbase + cc * 16 + m;
        float g = ag[rr][cc][r];
        float hv = (g / (1.0f + __expf(-g))) * au[rr][cc][r];
        H[(size_t)mr * FFD + fc] = f2bf(hv);
      }
}

// ---------------- FFN stage 2: out = (H Wd^T) * gate, scatter to (token,slot) ----------------
// block = 128 threads (4 wave32). Wave tile: 32 rows x 64 D cols.
// Per k-iter: 4 A loads + 8 B loads feed 8 WMMAs (1.5 b128 loads / WMMA).
__global__ void __launch_bounds__(128)
ffn2_kernel(const unsigned short* __restrict__ H,
            const unsigned short* __restrict__ Wdb,
            const float* __restrict__ rowWeight, const int* __restrict__ destIdx,
            const int* __restrict__ padOff, float* __restrict__ moeOut) {
  int rowBase = blockIdx.x * TILE_M;
  if (rowBase >= padOff[NEXP]) return;
  int e = 0;
#pragma unroll
  for (int i = 1; i < NEXP; ++i) if (rowBase >= padOff[i]) e = i;

  int lane = threadIdx.x & 31;
  int wave = threadIdx.x >> 5;
  int m = lane & 15;
  int h = lane >> 4;
  int dbase = blockIdx.y * 256 + wave * 64;         // this wave's 64 D cols

  const unsigned short* a0 = H + (size_t)(rowBase + m) * FFD;
  const unsigned short* a1 = H + (size_t)(rowBase + 16 + m) * FFD;
  const unsigned short* b[4];
#pragma unroll
  for (int cc = 0; cc < 4; ++cc)
    b[cc] = Wdb + ((size_t)e * DDIM + dbase + cc * 16 + m) * FFD;

  v8f acc[2][4];
#pragma unroll
  for (int rr = 0; rr < 2; ++rr)
#pragma unroll
    for (int cc = 0; cc < 4; ++cc)
      acc[rr][cc] = {0.f, 0.f, 0.f, 0.f, 0.f, 0.f, 0.f, 0.f};

  for (int k0 = 0; k0 < FFD; k0 += 32) {
    AFrag A0, A1, B[4];
    loadA(A0, a0, k0, h);
    loadA(A1, a1, k0, h);
#pragma unroll
    for (int cc = 0; cc < 4; ++cc) loadB(B[cc], b[cc], k0, h);
#pragma unroll
    for (int cc = 0; cc < 4; ++cc) {
      acc[0][cc] = wmma_bf16(A0, B[cc], acc[0][cc]);
      acc[1][cc] = wmma_bf16(A1, B[cc], acc[1][cc]);
    }
  }
#pragma unroll
  for (int rr = 0; rr < 2; ++rr)
#pragma unroll
    for (int r = 0; r < 8; ++r) {
      int mr = rowBase + rr * 16 + r + 8 * h;
      float w  = rowWeight[mr];
      size_t orow = (size_t)destIdx[mr] * DDIM;
#pragma unroll
      for (int cc = 0; cc < 4; ++cc)
        moeOut[orow + dbase + cc * 16 + m] = acc[rr][cc][r] * w;
    }
}

// ---------------- residual + LayerNorm ----------------
__global__ void __launch_bounds__(256)
ln_kernel(const float* __restrict__ x, const float* __restrict__ moeOut,
          const float* __restrict__ gamma, const float* __restrict__ beta,
          float* __restrict__ out) {
  __shared__ float ybuf[DDIM];
  __shared__ float red[256], red2[256];
  int tok = blockIdx.x;
  int t = threadIdx.x;
  float s = 0.0f, s2 = 0.0f;
  for (int d = t; d < DDIM; d += 256) {
    float y = x[(size_t)tok * DDIM + d]
            + moeOut[(size_t)(tok * 2 + 0) * DDIM + d]
            + moeOut[(size_t)(tok * 2 + 1) * DDIM + d];
    ybuf[d] = y; s += y; s2 += y * y;
  }
  red[t] = s; red2[t] = s2;
  __syncthreads();
  for (int st = 128; st > 0; st >>= 1) {
    if (t < st) { red[t] += red[t + st]; red2[t] += red2[t + st]; }
    __syncthreads();
  }
  float mean = red[0] / (float)DDIM;
  float var  = red2[0] / (float)DDIM - mean * mean;
  float rstd = rsqrtf(var + 1e-5f);
  for (int d = t; d < DDIM; d += 256)
    out[(size_t)tok * DDIM + d] = (ybuf[d] - mean) * rstd * gamma[d] + beta[d];
}

// ---------------- host launcher ----------------
extern "C" void kernel_launch(void* const* d_in, const int* in_sizes, int n_in,
                              void* d_out, int out_size, void* d_ws, size_t ws_size,
                              hipStream_t stream) {
  (void)in_sizes; (void)n_in; (void)out_size; (void)ws_size;
  const float* x     = (const float*)d_in[0];
  const float* gateW = (const float*)d_in[1];
  const float* Wg    = (const float*)d_in[2];
  const float* Wu    = (const float*)d_in[3];
  const float* Wd    = (const float*)d_in[4];
  const float* gamma = (const float*)d_in[5];
  const float* beta  = (const float*)d_in[6];
  float* out = (float*)d_out;

  char* ws = (char*)d_ws;
  size_t off = 0;
  auto alloc = [&](size_t bytes) -> char* {
    char* p = ws + off;
    off = (off + bytes + 255) & ~(size_t)255;
    return p;
  };
  unsigned short* xb   = (unsigned short*)alloc((size_t)NTOK * DDIM * 2);
  unsigned short* Wgb  = (unsigned short*)alloc((size_t)NEXP * FFD * DDIM * 2);
  unsigned short* Wub  = (unsigned short*)alloc((size_t)NEXP * FFD * DDIM * 2);
  unsigned short* Wdb  = (unsigned short*)alloc((size_t)NEXP * DDIM * FFD * 2);
  unsigned short* Hbuf = (unsigned short*)alloc((size_t)ROWS_PAD * FFD * 2);
  float* moeOut        = (float*)alloc((size_t)(2 * NTOK + 16) * DDIM * 4);
  int*   topkIdx       = (int*)alloc((size_t)NTOK * 2 * 4);
  float* topkW         = (float*)alloc((size_t)NTOK * 2 * 4);
  int*   rowToken      = (int*)alloc((size_t)ROWS_PAD * 4);
  float* rowWeight     = (float*)alloc((size_t)ROWS_PAD * 4);
  int*   destIdx       = (int*)alloc((size_t)ROWS_PAD * 4);
  int*   counts        = (int*)alloc(NEXP * 4);
  int*   padOff        = (int*)alloc((NEXP + 1) * 4);
  int*   cursor        = (int*)alloc(NEXP * 4);
  float* sumP          = (float*)alloc(NEXP * 4);
  float* sumZ          = (float*)alloc(4);

  // 1. init
  init_kernel<<<(ROWS_PAD + 255) / 256, 256, 0, stream>>>(counts, sumP, sumZ,
                                                          rowToken, rowWeight, destIdx);
  // 2. weight conversion fp32 -> bf16 (once; reused ~128x by the GEMMs)
  const int n4 = NEXP * FFD * DDIM / 4;
  wcvt_kernel<<<(n4 + 255) / 256, 256, 0, stream>>>(Wg, Wgb, n4);
  wcvt_kernel<<<(n4 + 255) / 256, 256, 0, stream>>>(Wu, Wub, n4);
  wcvt_kernel<<<(n4 + 255) / 256, 256, 0, stream>>>(Wd, Wdb, n4);
  // 3. router (+ x -> bf16)
  router_kernel<<<NTOK, 256, 0, stream>>>(x, gateW, xb, topkIdx, topkW,
                                          counts, sumP, sumZ);
  // 4. padded scan, 5. scatter, 6. aux scalar
  scan_kernel<<<1, 32, 0, stream>>>(counts, padOff, cursor);
  scatter_kernel<<<(NTOK + 255) / 256, 256, 0, stream>>>(topkIdx, topkW, cursor,
                                                         rowToken, rowWeight, destIdx);
  aux_kernel<<<1, 32, 0, stream>>>(counts, sumP, sumZ, out + (size_t)NTOK * DDIM);
  // 7. FFN1: 32-row x 128-FF blocks (4 waves x 32x32 dual-matrix tiles)
  ffn1_kernel<<<dim3(ROW_TILES, FFD / 128), 128, 0, stream>>>(xb, Wgb, Wub,
                                                              rowToken, padOff, Hbuf);
  // 8. FFN2: 32-row x 256-D blocks (4 waves x 32x64 tiles), gated scatter
  ffn2_kernel<<<dim3(ROW_TILES, DDIM / 256), 128, 0, stream>>>(Hbuf, Wdb, rowWeight,
                                                               destIdx, padOff, moeOut);
  // 9. residual + LayerNorm
  ln_kernel<<<NTOK, 256, 0, stream>>>(x, moeOut, gamma, beta, out);
}
